// CPHashLayer_53919019434153
// MI455X (gfx1250) — matrix operand order repaired and checked by reading, not compiled, not run on hardware
//
#include <hip/hip_runtime.h>
#include <hip/hip_bf16.h>

// CP-ALS (rank 32, 20 iters) + MLP hash head, fused per-sample.
// One workgroup (256 threads = 8 wave32) per batch sample.
// MTTKRP inner loops run on v_wmma_f32_16x16x32_bf16 (bf16 in, f32 accum).

typedef __attribute__((ext_vector_type(16))) __bf16 v16bf;
typedef __attribute__((ext_vector_type(8)))  float  v8f;

#define RANK   32
#define CC     128          // mode-0 dim (channels)
#define HH     56           // mode-1 dim
#define WW     56           // mode-2 dim
#define JK     (HH * WW)    // 3136 = 98 * 32
#define IK     (CC * WW)    // 7168 = 224 * 32
#define NITERS 20
#define RIDGE  1e-6f
#define NTHR   256

__device__ __forceinline__ v8f wmma_bf16(v16bf a, v16bf b, v8f c) {
  // D = A(16x32 bf16) x B(32x16 bf16) + C(16x16 f32)
  return __builtin_amdgcn_wmma_f32_16x16x32_bf16(
      /*neg_a=*/false, a, /*neg_b=*/false, b,
      /*c_mod=*/(short)0, c, /*reuse_a=*/false, /*reuse_b=*/false);
}

// V[r1][r2] = (F1^T F1)[r1][r2] * (F2^T F2)[r1][r2] + ridge*I
__device__ __forceinline__ void compute_V(float (*V)[RANK],
                                          const float (*F1)[RANK], int n1,
                                          const float (*F2)[RANK], int n2) {
#pragma unroll
  for (int jj = 0; jj < 4; ++jj) {
    int id = threadIdx.x + jj * NTHR;       // 0..1023 => all 32x32 entries
    int r1 = id >> 5, r2 = id & 31;
    float g1 = 0.f, g2 = 0.f;
    for (int m = 0; m < n1; ++m) g1 += F1[m][r1] * F1[m][r2];
    for (int m = 0; m < n2; ++m) g2 += F2[m][r1] * F2[m][r2];
    V[r1][r2] = g1 * g2 + ((r1 == r2) ? RIDGE : 0.f);
  }
}

// In-place lower Cholesky of V, then per-row solve V x = M[m]^T, write Fout[m] = x.
__device__ __forceinline__ void chol_solve_write(float (*V)[RANK],
                                                 const float (*Mm)[RANK],
                                                 float (*Fout)[RANK], int mdim) {
  for (int t = 0; t < RANK; ++t) {
    if (threadIdx.x == 0) V[t][t] = sqrtf(V[t][t]);
    __syncthreads();
    if (threadIdx.x > t && threadIdx.x < RANK) V[threadIdx.x][t] /= V[t][t];
    __syncthreads();
#pragma unroll
    for (int jj = 0; jj < 4; ++jj) {
      int id = threadIdx.x + jj * NTHR;
      int r = id >> 5, c = id & 31;
      if (r > t && c > t && c <= r) V[r][c] -= V[r][t] * V[c][t];
    }
    __syncthreads();
  }
  int m = threadIdx.x;
  if (m < mdim) {
    float y[RANK];
#pragma unroll
    for (int c = 0; c < RANK; ++c) {        // forward: L y = rhs
      float s = Mm[m][c];
#pragma unroll
      for (int j = 0; j < c; ++j) s -= V[c][j] * y[j];
      y[c] = s / V[c][c];
    }
#pragma unroll
    for (int c = RANK - 1; c >= 0; --c) {   // backward: L^T x = y
      float s = y[c];
#pragma unroll
      for (int j = c + 1; j < RANK; ++j) s -= V[j][c] * y[j];
      y[c] = s / V[c][c];
    }
#pragma unroll
    for (int r = 0; r < RANK; ++r) Fout[m][r] = y[r];
  }
  __syncthreads();
}

// ---- mode-0 MTTKRP: M0[i,r] = sum_{j,k} T[i,j,k] B[j,r] C[k,r] ----
// Unfolding rows (fixed i) are contiguous memory -> vectorized A-fragment loads.
__device__ __forceinline__ void mttkrp_mode0(const float* __restrict__ T,
                                             float (*Mout)[RANK],
                                             const float (*FB)[RANK],
                                             const float (*FC)[RANK],
                                             __bf16 (*krt)[32]) {
  const int tid = threadIdx.x, lane = tid & 31, wave = tid >> 5;
  const int sel = lane >> 4;                      // K-half select (WMMA A layout)
  const int row = wave * 16 + (lane & 15);        // 8 M-tiles x 16 rows = 128
  const float* Trow = T + (size_t)row * JK;
  v8f acc0 = {0, 0, 0, 0, 0, 0, 0, 0};
  v8f acc1 = {0, 0, 0, 0, 0, 0, 0, 0};
  for (int p = 0; p < JK; p += 32) {
    // Build Khatri-Rao tile transposed: krt[n][kk] = B[(p+kk)/56][n]*C[(p+kk)%56][n]
#pragma unroll
    for (int jj = 0; jj < 4; ++jj) {
      int id = tid + jj * NTHR;
      int kk = id >> 5, n = id & 31;
      int idx = p + kk;
      int jv = idx / WW, kv = idx - jv * WW;
      krt[n][kk] = (__bf16)(FB[jv][n] * FC[kv][n]);
    }
    __syncthreads();
    // A fragment straight from global (rows contiguous): lanes 0-15 get K 0-15,
    // lanes 16-31 get K 16-31 (contiguous order, matching the B fragment).
    const float4* q0 = (const float4*)(Trow + p + sel * 16);
    const float4* q1 = (const float4*)(Trow + p + sel * 16 + 8);
    float4 x0 = q0[0], x1 = q0[1], y0 = q1[0], y1 = q1[1];
    if (p + 32 < JK) __builtin_prefetch(Trow + p + 32, 0, 1);  // global_prefetch_b8
    v16bf a;
    a[0]  = (__bf16)x0.x; a[1]  = (__bf16)x0.y; a[2]  = (__bf16)x0.z; a[3]  = (__bf16)x0.w;
    a[4]  = (__bf16)x1.x; a[5]  = (__bf16)x1.y; a[6]  = (__bf16)x1.z; a[7]  = (__bf16)x1.w;
    a[8]  = (__bf16)y0.x; a[9]  = (__bf16)y0.y; a[10] = (__bf16)y0.z; a[11] = (__bf16)y0.w;
    a[12] = (__bf16)y1.x; a[13] = (__bf16)y1.y; a[14] = (__bf16)y1.z; a[15] = (__bf16)y1.w;
    // B fragments: lane holds one column n, 16 contiguous halves of its K-half
    v16bf bf0 = *(const v16bf*)&krt[(lane & 15)][sel * 16];
    v16bf bf1 = *(const v16bf*)&krt[16 + (lane & 15)][sel * 16];
    acc0 = wmma_bf16(a, bf0, acc0);
    acc1 = wmma_bf16(a, bf1, acc1);
    __syncthreads();
  }
#pragma unroll
  for (int v = 0; v < 8; ++v) {            // D layout: row = v + 8*(lane>>4)
    int r = wave * 16 + v + sel * 8;
    Mout[r][(lane & 15)]      = acc0[v];
    Mout[r][16 + (lane & 15)] = acc1[v];
  }
}

// ---- modes 1/2 (strided unfoldings, per-element gather for the A fragment) ----
// MODE==1: M1[j,r] = sum_{i,k} T[i,j,k] A[i,r] C[k,r]   (idx = i*56+k)
// MODE==2: M2[k,r] = sum_{i,j} T[i,j,k] A[i,r] B[j,r]   (idx = i*56+j)
template <int MODE>
__device__ __forceinline__ void mttkrp_gather(const float* __restrict__ T,
                                              float (*Mout)[RANK],
                                              const float (*F1)[RANK],
                                              const float (*F2)[RANK],
                                              __bf16 (*krt)[32]) {
  const int tid = threadIdx.x, lane = tid & 31, wave = tid >> 5;
  const int mt = wave & 3, nt = wave >> 2;        // 4 M-tiles (56->64) x 2 N-tiles
  const int sel = lane >> 4;
  const int r = mt * 16 + (lane & 15);
  const bool valid = (r < HH);
  v8f acc = {0, 0, 0, 0, 0, 0, 0, 0};
  for (int p = 0; p < IK; p += 32) {
#pragma unroll
    for (int jj = 0; jj < 4; ++jj) {
      int id = tid + jj * NTHR;
      int kk = id >> 5, n = id & 31;
      int idx = p + kk;
      int iv = idx / WW, qv = idx - iv * WW;
      krt[n][kk] = (__bf16)(F1[iv][n] * F2[qv][n]);
    }
    __syncthreads();
    v16bf a;
#pragma unroll
    for (int h = 0; h < 16; ++h) {
      int k = sel * 16 + h;
      int idx = p + k;
      int iv = idx / WW, qv = idx - iv * WW;
      size_t off = (MODE == 1) ? ((size_t)iv * JK + (size_t)r * WW + qv)
                               : ((size_t)iv * JK + (size_t)qv * WW + r);
      a[h] = valid ? (__bf16)T[off] : (__bf16)0.f;
    }
    v16bf bf = *(const v16bf*)&krt[nt * 16 + (lane & 15)][sel * 16];
    acc = wmma_bf16(a, bf, acc);
    __syncthreads();
  }
#pragma unroll
  for (int v = 0; v < 8; ++v) {
    int rr = mt * 16 + v + sel * 8;
    if (rr < HH) Mout[rr][nt * 16 + (lane & 15)] = acc[v];
  }
}

__global__ __launch_bounds__(NTHR, 1)
void cp_als_hash_kernel(const float* __restrict__ x,
                        const float* __restrict__ W1, const float* __restrict__ b1,
                        const float* __restrict__ W2, const float* __restrict__ b2,
                        const float* __restrict__ A0, const float* __restrict__ B0,
                        const float* __restrict__ C0,
                        float* __restrict__ out, int Bsz) {
  __shared__ float sA[CC][RANK];                      // 16 KB
  __shared__ float sB[HH][RANK];                      // 7 KB
  __shared__ float sC[WW][RANK];                      // 7 KB
  __shared__ float sM[CC][RANK];                      // 16 KB MTTKRP result
  __shared__ float sV[RANK][RANK];                    // 4 KB normal matrix
  __shared__ __align__(64) __bf16 krt[RANK][32];      // 2 KB Khatri-Rao tile (B^T)
  __shared__ float sfeat[3 * RANK];
  __shared__ float sh[NTHR];

  const int tid = threadIdx.x;
  const int b = blockIdx.x;
  const float* T = x + (size_t)b * (CC * JK);

  // Init factors from A0/B0/C0
  for (int i = tid; i < CC * RANK; i += NTHR) (&sA[0][0])[i] = A0[(size_t)b * CC * RANK + i];
  for (int i = tid; i < HH * RANK; i += NTHR) (&sB[0][0])[i] = B0[(size_t)b * HH * RANK + i];
  for (int i = tid; i < WW * RANK; i += NTHR) (&sC[0][0])[i] = C0[(size_t)b * WW * RANK + i];
  __syncthreads();

  for (int it = 0; it < NITERS; ++it) {
    // mode 0: update A
    mttkrp_mode0(T, sM, sB, sC, krt);
    __syncthreads();
    compute_V(sV, sB, HH, sC, WW);
    __syncthreads();
    chol_solve_write(sV, sM, sA, CC);
    // mode 1: update B
    mttkrp_gather<1>(T, sM, sA, sC, krt);
    __syncthreads();
    compute_V(sV, sA, CC, sC, WW);
    __syncthreads();
    chol_solve_write(sV, sM, sB, HH);
    // mode 2: update C
    mttkrp_gather<2>(T, sM, sA, sB, krt);
    __syncthreads();
    compute_V(sV, sA, CC, sB, HH);
    __syncthreads();
    chol_solve_write(sV, sM, sC, WW);
  }

  // Features: per-mode factor means -> (3*RANK)
  if (tid < 3 * RANK) {
    int grp = tid >> 5, r = tid & 31;
    float s = 0.f;
    if (grp == 0)      { for (int m = 0; m < CC; ++m) s += sA[m][r]; s *= (1.f / CC); }
    else if (grp == 1) { for (int m = 0; m < HH; ++m) s += sB[m][r]; s *= (1.f / HH); }
    else               { for (int m = 0; m < WW; ++m) s += sC[m][r]; s *= (1.f / WW); }
    sfeat[tid] = s;
  }
  __syncthreads();

  // MLP head: relu(feats @ W1 + b1) @ W2 + b2
  {
    float acc = b1[tid];
    for (int f = 0; f < 3 * RANK; ++f) acc += sfeat[f] * W1[f * 256 + tid];
    sh[tid] = fmaxf(acc, 0.f);
  }
  __syncthreads();
  if (tid < 128) {
    float acc = b2[tid];
    for (int j = 0; j < 256; ++j) acc += sh[j] * W2[j * 128 + tid];
    float sg = (acc > 0.f) ? 1.f : ((acc < 0.f) ? -1.f : 0.f);
    out[(size_t)b * 128 + tid] = sg;                          // binary_hash
    out[(size_t)Bsz * 128 + (size_t)b * 128 + tid] = acc;     // hash_logits
  }
}

extern "C" void kernel_launch(void* const* d_in, const int* in_sizes, int n_in,
                              void* d_out, int out_size, void* d_ws, size_t ws_size,
                              hipStream_t stream) {
  const float* x  = (const float*)d_in[0];
  const float* W1 = (const float*)d_in[1];
  const float* b1 = (const float*)d_in[2];
  const float* W2 = (const float*)d_in[3];
  const float* b2 = (const float*)d_in[4];
  const float* A0 = (const float*)d_in[5];
  const float* B0 = (const float*)d_in[6];
  const float* C0 = (const float*)d_in[7];
  float* out = (float*)d_out;
  int Bsz = in_sizes[0] / (CC * HH * WW);
  hipLaunchKernelGGL(cp_als_hash_kernel, dim3(Bsz), dim3(NTHR), 0, stream,
                     x, W1, b1, W2, b2, A0, B0, C0, out, Bsz);
}
